// VectorQuantizer_40398462386425
// MI455X (gfx1250) — matrix operand order repaired and checked by reading, not compiled, not run on hardware
//
#include <hip/hip_runtime.h>
#include <hip/hip_bf16.h>

// ---------------------------------------------------------------------------
// VQ (vector quantizer) for MI455X / gfx1250.
// z: [16,256,32,32] f32  -> N=16384 tokens of dim 256
// W: [16384,256] f32 codebook
// out: z_q (4194304 f32) | loss (1 f32) | idx (16384 int32 bit-patterns)
//
// Distance GEMM (137 GFLOP) dominates -> f16 WMMA (v_wmma_f32_16x16x32_f16)
// with fused argmin epilogue. Codebook tiles are staged into LDS by the
// Tensor Data Mover (tensor_load_to_lds, TENSORcnt) with hardware padding,
// double-buffered so DMA overlaps the WMMA pipe.
// ---------------------------------------------------------------------------

typedef __attribute__((ext_vector_type(16))) _Float16 v16h;
typedef __attribute__((ext_vector_type(8)))  _Float16 v8h;
typedef __attribute__((ext_vector_type(8)))  float    v8f;
typedef __attribute__((ext_vector_type(4)))  unsigned v4u;
typedef __attribute__((ext_vector_type(8)))  int      v8i;
typedef __attribute__((ext_vector_type(4)))  int      v4i;

#define N_TOK   16384
#define N_EMB   16384
#define DIM     256
#define KSPLIT  4
#define KSEG    (N_EMB / KSPLIT)      // 4096 codes per K-segment
#define ROWS_PER_BLOCK 128            // 8 waves x 16 rows
#define KTILE   64                    // codebook rows staged in LDS per iter
#define LDSH    264                   // padded LDS row stride (halfs): 528B, 16B aligned, bank-skewed

#if __has_builtin(__builtin_amdgcn_tensor_load_to_lds)
#define HAVE_TDM 1
#else
#define HAVE_TDM 0
#endif

// ---- prep: codebook -> f16 + row norms -----------------------------------
__global__ void prep_w(const float* __restrict__ W, _Float16* __restrict__ wf16,
                       float* __restrict__ wn) {
  int row = blockIdx.x;
  int t = threadIdx.x;
  float v = W[(size_t)row * DIM + t];
  wf16[(size_t)row * DIM + t] = (_Float16)v;
  __shared__ float red[256];
  red[t] = v * v;
  __syncthreads();
  for (int s = 128; s > 0; s >>= 1) { if (t < s) red[t] += red[t + s]; __syncthreads(); }
  if (t == 0) wn[row] = red[0];
}

// ---- prep: z [B,C,H,W] -> zf16 [N,DIM] row-major (LDS tile transpose) ----
__global__ void prep_z(const float* __restrict__ z, _Float16* __restrict__ zf16) {
  __shared__ float tile[32][33];
  int b = blockIdx.x, c0 = blockIdx.y * 32, hw0 = blockIdx.z * 32;
  int tx = threadIdx.x, ty = threadIdx.y;         // 32 x 8
  #pragma unroll
  for (int i = 0; i < 4; i++) {
    int c = c0 + ty + 8 * i;
    tile[ty + 8 * i][tx] = z[((size_t)(b * DIM) + c) * 1024 + hw0 + tx];
  }
  __syncthreads();
  #pragma unroll
  for (int i = 0; i < 4; i++) {
    int r = ty + 8 * i;                           // hw offset
    zf16[(size_t)(b * 1024 + hw0 + r) * DIM + c0 + tx] = (_Float16)tile[tx][r];
  }
}

#if HAVE_TDM
// Issue one TDM descriptor: copy KTILE*DIM halfs (32KB, contiguous in global)
// into LDS at lds_byte, inserting 16B of padding after every 512B so the LDS
// row stride is LDSH halfs (bank-skewed). D# per ISA ch.8:
//   group0: count=1 | lds_addr | global_addr[56:0] | type=2
//   group1: data_size=8B, pad_enable, pad_interval=128 DW, pad_amount=4 DW,
//           1-D tile: tile_dim0 = 4096 x 8B units.
__device__ __forceinline__ void tdm_stage(unsigned lds_byte, const _Float16* gsrc) {
  unsigned long long ga = (unsigned long long)(uintptr_t)gsrc;
  v4u g0;
  g0.x = 1u;                                               // count=1, user desc
  g0.y = lds_byte;                                         // lds_addr
  g0.z = (unsigned)(ga & 0xffffffffu);                     // global_addr[31:0]
  g0.w = (unsigned)((ga >> 32) & 0x01ffffffu) | 0x80000000u; // addr[56:32] | type=2
  v8i g1;
  g1[0] = (3 << 16)      // data_size = 8B
        | (1 << 20)      // pad_enable
        | (6 << 22)      // pad_interval: 128 DWORDs = 512B
        | (3 << 25);     // pad_amount: 4 DWORDs = 16B
  unsigned td0 = 1u << 30;                                 // huge tensor_dim0 (tile never OOB)
  g1[1] = (int)((td0 & 0xffffu) << 16);                    // [63:48] = tensor_dim0[15:0]
  g1[2] = (int)((td0 >> 16) & 0xffffu);                    // [79:64] = tensor_dim0[31:16]
  g1[3] = (int)(4096u << 16);                              // [127:112] = tile_dim0 = 4096 units
  g1[4] = 0;                                               // tile_dim1/2 unused (1-D)
  g1[5] = 0; g1[6] = 0; g1[7] = 0;                         // strides unused
  v4i z4 = {0, 0, 0, 0};
  v8i z8 = {0, 0, 0, 0, 0, 0, 0, 0};
  // 6-arg form (clang-23 / therock-10.0 headers): (g0, g1, g2, g3, g4, cpol)
  __builtin_amdgcn_tensor_load_to_lds(g0, g1, z4, z4, z8, 0);
}
#endif

// ---- main: WMMA GEMM + fused argmin --------------------------------------
// grid (N_TOK/ROWS_PER_BLOCK, KSPLIT), block 256 (8 waves)
__global__ __launch_bounds__(256) void vq_argmin(
    const _Float16* __restrict__ zf16, const _Float16* __restrict__ wf16,
    const float* __restrict__ wn, float* __restrict__ pval, int* __restrict__ pidx) {
  __shared__ __align__(128) _Float16 lw[2][KTILE * LDSH];  // 2 x 33.8 KB double buffer

  int blk  = blockIdx.x;
  int seg  = blockIdx.y;
  int t    = threadIdx.x;
  int wave = t >> 5;
  int lane = t & 31;
  int m    = lane & 15;                           // A-row / B-col within 16x16 tile
  int h    = lane >> 4;                           // half-wave group
  int n0   = blk * ROWS_PER_BLOCK + wave * 16;

  // Preload this wave's 16 A-rows (16x256 f16) into VGPRs for the whole K loop.
  // ISA A-layout (16-bit 16x32): VGPR0-3 = K h*8..h*8+7, VGPR4-7 = K 16+h*8..+7.
  v16h a[8];
  const _Float16* arow = zf16 + (size_t)(n0 + m) * DIM;
  #pragma unroll
  for (int cc = 0; cc < 8; cc++) {
    v8h lo = *(const v8h*)(arow + cc * 32 + h * 8);
    v8h hi = *(const v8h*)(arow + cc * 32 + 16 + h * 8);
    a[cc] = __builtin_shufflevector(lo, hi, 0,1,2,3,4,5,6,7,8,9,10,11,12,13,14,15);
  }

  float best[8]; int bidx[8];
  #pragma unroll
  for (int r = 0; r < 8; r++) { best[r] = 3.4e38f; bidx[r] = 0; }

  int kbase = seg * KSEG;
  const int NIT = KSEG / KTILE;

#if HAVE_TDM
  // Prologue: wave 0 launches the DMA for the first tile.
  if (wave == 0) tdm_stage((unsigned)(uintptr_t)&lw[0][0], wf16 + (size_t)kbase * DIM);
#endif

  for (int it = 0; it < NIT; it++) {
    int k0  = kbase + it * KTILE;
    int cur = it & 1;

#if HAVE_TDM
    if (wave == 0) __builtin_amdgcn_s_wait_tensorcnt((short)0); // tile[cur] resident
    __syncthreads();                                            // publish to all waves
    if (wave == 0 && it + 1 < NIT)                              // prefetch next tile
      tdm_stage((unsigned)(uintptr_t)&lw[1 - cur][0],
                wf16 + (size_t)(k0 + KTILE) * DIM);
#else
    cur = 0;
    __syncthreads();
    // Fallback: cooperative staging (16B chunks, coalesced).
    #pragma unroll
    for (int q = 0; q < 8; q++) {
      int id  = t + 256 * q;                      // 0..2047
      int row = id >> 5;                          // 0..63
      int ch  = id & 31;                          // 16B chunk in row
      *(v8h*)(&lw[0][0] + row * LDSH + ch * 8) =
          *(const v8h*)(wf16 + (size_t)(k0 + row) * DIM + ch * 8);
    }
    __syncthreads();
#endif

    const _Float16* lwc = &lw[cur][0];
    #pragma unroll
    for (int j0 = 0; j0 < KTILE; j0 += 16) {
      v8f c = {};
      // ISA B-layout (16-bit): half-wave h holds K h*8..h*8+7, 16+h*8..+7;
      // lane's column = m.
      const _Float16* brow = lwc + (j0 + m) * LDSH;
      #pragma unroll
      for (int cc = 0; cc < 8; cc++) {
        v8h lo = *(const v8h*)(brow + cc * 32 + h * 8);
        v8h hi = *(const v8h*)(brow + cc * 32 + 16 + h * 8);
        v16h bf = __builtin_shufflevector(lo, hi, 0,1,2,3,4,5,6,7,8,9,10,11,12,13,14,15);
        c = __builtin_amdgcn_wmma_f32_16x16x32_f16(false, a[cc], false, bf,
                                                   (short)0, c, false, false);
      }
      // Fused epilogue: score = ||e||^2 - 2 z.e ; running argmin per row.
      int kcol = k0 + j0 + m;
      float w2 = wn[kcol];
      #pragma unroll
      for (int r = 0; r < 8; r++) {               // row M = 8*h + r (C/D layout)
        float s = w2 - 2.0f * c[r];
        if (s < best[r]) { best[r] = s; bidx[r] = kcol; }
      }
    }
  }

  // Argmin across the 16 lanes that share each row (first-min tie-break).
  #pragma unroll
  for (int r = 0; r < 8; r++) {
    #pragma unroll
    for (int off = 1; off < 16; off <<= 1) {
      float ov = __shfl_xor(best[r], off, 32);
      int   oi = __shfl_xor(bidx[r], off, 32);
      if (ov < best[r] || (ov == best[r] && oi < bidx[r])) { best[r] = ov; bidx[r] = oi; }
    }
    if (m == 0) {
      int row = n0 + 8 * h + r;
      pval[(size_t)seg * N_TOK + row] = best[r];
      pidx[(size_t)seg * N_TOK + row] = bidx[r];
    }
  }
}

// ---- merge K segments (deterministic, first-min tie-break) ---------------
__global__ void merge_argmin(const float* __restrict__ pval, const int* __restrict__ pidx,
                             int* __restrict__ idx_out) {
  int n = blockIdx.x * 256 + threadIdx.x;
  float bv = pval[n]; int bi = pidx[n];
  for (int s = 1; s < KSPLIT; s++) {
    float v = pval[(size_t)s * N_TOK + n];
    int   i = pidx[(size_t)s * N_TOK + n];
    if (v < bv || (v == bv && i < bi)) { bv = v; bi = i; }
  }
  idx_out[n] = bi;
}

// ---- gather z_q = W[idx] back to [B,C,H,W] + loss partials ---------------
__global__ void gather_zq(const float* __restrict__ z, const float* __restrict__ W,
                          const int* __restrict__ idx, float* __restrict__ zq,
                          float* __restrict__ partial) {
  __shared__ int sidx[32];
  __shared__ float red[256];
  int b = blockIdx.x, c0 = blockIdx.y * 32, hw0 = blockIdx.z * 32;
  int tx = threadIdx.x, ty = threadIdx.y;
  int t = ty * 32 + tx;
  if (t < 32) sidx[t] = idx[b * 1024 + hw0 + t];
  __syncthreads();
  float acc = 0.f;
  #pragma unroll
  for (int i = 0; i < 4; i++) {
    int c = c0 + ty + 8 * i;
    size_t off = ((size_t)(b * DIM) + c) * 1024 + hw0 + tx;  // coalesced in tx
    float zv = z[off];
    float wv = W[(size_t)sidx[tx] * DIM + c];                // L2-resident gather
    zq[off] = wv;
    float d = wv - zv;
    acc += d * d;
  }
  red[t] = acc;
  __syncthreads();
  for (int s = 128; s > 0; s >>= 1) { if (t < s) red[t] += red[t + s]; __syncthreads(); }
  if (t == 0)
    partial[(blockIdx.x * gridDim.y + blockIdx.y) * gridDim.z + blockIdx.z] = red[0];
}

// ---- loss = 1.25 * mean((z_q - z)^2), fixed-order reduction --------------
__global__ void loss_final(const float* __restrict__ partial, float* __restrict__ loss_out) {
  __shared__ float red[256];
  int t = threadIdx.x;
  float acc = 0.f;
  for (int i = 0; i < 16; i++) acc += partial[t * 16 + i];
  red[t] = acc;
  __syncthreads();
  for (int s = 128; s > 0; s >>= 1) { if (t < s) red[t] += red[t + s]; __syncthreads(); }
  if (t == 0) *loss_out = 1.25f * red[0] / (float)((size_t)N_TOK * DIM);
}

extern "C" void kernel_launch(void* const* d_in, const int* in_sizes, int n_in,
                              void* d_out, int out_size, void* d_ws, size_t ws_size,
                              hipStream_t stream) {
  (void)in_sizes; (void)n_in; (void)out_size; (void)ws_size;
  const float* z = (const float*)d_in[0];
  const float* W = (const float*)d_in[1];
  float* out = (float*)d_out;
  char* ws = (char*)d_ws;

  // workspace layout (~16.6 MB)
  _Float16* wf16   = (_Float16*)(ws);                  // 8 MB
  _Float16* zf16   = (_Float16*)(ws + 8388608);        // 8 MB
  float*    wn     = (float*)(ws + 16777216);          // 64 KB
  float*    pval   = (float*)(ws + 16842752);          // 256 KB
  int*      pidx   = (int*)(ws + 17104896);            // 256 KB
  float*    partial= (float*)(ws + 17367040);          // 16 KB

  float* zq   = out;                                   // 4194304 f32
  float* loss = out + 4194304;                         // 1 f32
  int*   idxo = (int*)(out + 4194305);                 // 16384 int32 bits

  prep_w   <<<dim3(N_EMB),               dim3(256),    0, stream>>>(W, wf16, wn);
  prep_z   <<<dim3(16, 8, 32),           dim3(32, 8),  0, stream>>>(z, zf16);
  vq_argmin<<<dim3(N_TOK / ROWS_PER_BLOCK, KSPLIT), dim3(256), 0, stream>>>(zf16, wf16, wn, pval, pidx);
  merge_argmin<<<dim3(N_TOK / 256),      dim3(256),    0, stream>>>(pval, pidx, idxo);
  gather_zq<<<dim3(16, 8, 32),           dim3(32, 8),  0, stream>>>(z, W, idxo, zq, partial);
  loss_final<<<dim3(1),                  dim3(256),    0, stream>>>(partial, loss);
}
